// SimpleNet_35313221107878
// MI455X (gfx1250) — matrix operand order repaired and checked by reading, not compiled, not run on hardware
//
#include <hip/hip_runtime.h>

#define NVv 50000L
#define NCc 50000L
#define Ee  800000L

typedef __bf16 bf16;
typedef __attribute__((ext_vector_type(2)))  __bf16 bf16x2;
typedef __attribute__((ext_vector_type(16))) __bf16 bf16x16;
typedef __attribute__((ext_vector_type(8)))  float  f32x8;

__device__ __forceinline__ f32x8 wmma_bf16f32(bf16x16 a, bf16x16 b, f32x8 c) {
  // D = A(16x32 bf16) * B(32x16 bf16) + C(16x16 f32)
  return __builtin_amdgcn_wmma_f32_16x16x32_bf16(false, a, false, b, (short)0, c, false, false);
}

__device__ __forceinline__ float actf(float v, int act) {
  if (act == 1) return fmaxf(v, 0.f);
  if (act == 2) return 1.f / (1.f + expf(-v));
  return v;
}

// ------------------------------ utility ------------------------------
__global__ void k_zero(float* __restrict__ p, long n) {
  long i = (long)blockIdx.x * blockDim.x + threadIdx.x;
  long st = (long)gridDim.x * blockDim.x;
  for (; i < n; i += st) p[i] = 0.f;
}

// --------- weight prep: W[K,nout] f32 -> padded bf16 B-fragment layout ------
// Fragment element (s, colsub, lane, i) holds W[s*32 + (i>=8?16:0) + (lane>=16?8:0) + (i&7)]
//                                             [colsub*16 + (lane&15)]
__global__ void k_prep_w(const float* __restrict__ W, int K, int nout,
                         bf16* __restrict__ F, int Kpad)
{
  long n = (long)Kpad * 64;
  long i = (long)blockIdx.x * blockDim.x + threadIdx.x;
  long stp = (long)gridDim.x * blockDim.x;
  for (; i < n; i += stp) {
    int kg = (int)(i >> 6), c = (int)(i & 63);
    float v = (kg < K && c < nout) ? W[(long)kg * nout + c] : 0.f;
    int s = kg >> 5, kl = kg & 31;
    int ln = (c & 15) + (((kl & 15) >= 8) ? 16 : 0);
    int ii = ((kl >= 16) ? 8 : 0) + (kl & 7);
    F[(((long)(s * 4 + (c >> 4)) * 32 + ln) * 16) + ii] = (bf16)v;
  }
}

// ------------------- GEMM: Y = act(X @ W + b), K compile-time --------------
// K is a multiple of 32 (inputs/weights zero-padded at prep time).
// A-tile: rows clamped to M-1 (garbage rows never stored). No k-bounds at all.
// B fragments come pre-swizzled from global (one 32B load/wave/k-step).
template <int K>
__global__ void k_gemm_t(const float* __restrict__ X, int ldx, long M,
                         const bf16* __restrict__ Wf, const float* __restrict__ B,
                         int nout, float* __restrict__ Y, int ldy, int coff, int act)
{
  __shared__ __align__(32) bf16 sA[2][32][16];   // [rowsub][lane][elem]
  long rowbase = (long)blockIdx.x * 32;
  int t = threadIdx.x;
  int lane = t & 31, wv = t >> 5;
  int rowsub = wv >> 2, colsub = wv & 3;
  f32x8 acc = {0.f,0.f,0.f,0.f,0.f,0.f,0.f,0.f};
#pragma unroll
  for (int s = 0; s < K / 32; ++s) {
    // stage X tile (32 rows x 32 k): 512 bf16x2 pairs, branch-free, no selects
#pragma unroll
    for (int pp = 0; pp < 2; ++pp) {
      int pid = t + pp * 256;            // 0..511
      int r = pid >> 4;                  // 0..31
      int k = (pid & 15) * 2;            // even 0..30
      long row = rowbase + r;
      long rc = row < M ? row : (M - 1);
      const float* xp = &X[rc * (long)ldx + s * 32 + k];
      bf16x2 pr; pr.x = (bf16)xp[0]; pr.y = (bf16)xp[1];
      int ln = (r & 15) + (((k & 15) >= 8) ? 16 : 0);
      int ii = ((k >= 16) ? 8 : 0) + (k & 7);       // even
      *(bf16x2*)&sA[r >> 4][ln][ii] = pr;
    }
    __syncthreads();
    bf16x16 fa = *(const bf16x16*)&sA[rowsub][lane][0];      // 2x ds_load_b128
    bf16x16 fb = ((const bf16x16*)Wf)[(s * 4 + colsub) * 32 + lane];  // global 32B
    acc = wmma_bf16f32(fa, fb, acc);
    __syncthreads();
  }
  int ncol = colsub * 16 + (lane & 15);
  int mb = rowsub * 16 + ((lane < 16) ? 0 : 8);
  float bias = B[ncol < nout ? ncol : 0];
#pragma unroll
  for (int r = 0; r < 8; ++r) {
    long row = rowbase + mb + r;
    if (row < M && ncol < nout)
      Y[row * (long)ldy + coff + ncol] = actf(acc[r] + bias, act);
  }
}

// ---- fused 2-layer MLP with tiny first layer (kin<=2, dmid=64) ----
// h = relu(X@W1+b1) built directly in LDS (swizzled); Y = act(h@W2f+b2)
__global__ void k_fused_mlp2(const float* __restrict__ X, int kin, long M,
                             const float* __restrict__ W1, const float* __restrict__ B1,
                             const bf16* __restrict__ W2f, const float* __restrict__ B2,
                             int nout, float* __restrict__ Y, int ldy, int coff, int act)
{
  __shared__ __align__(32) bf16 sA[2][32][32];   // [rowsub][lane][kstep*16+i]
  long rowbase = (long)blockIdx.x * 32;
  int t = threadIdx.x, lane = t & 31, wv = t >> 5;
  int rowsub = wv >> 2, colsub = wv & 3;
  // h1 tile 32x64: 1024 pairs along c, branch-free (rows clamped, not zeroed)
#pragma unroll
  for (int pp = 0; pp < 4; ++pp) {
    int pid = t + pp * 256;              // 0..1023
    int r = pid >> 5;
    int c = (pid & 31) * 2;              // even 0..62
    long row = rowbase + r;
    long rc = row < M ? row : (M - 1);
    float v0 = B1[c], v1 = B1[c + 1];
    for (int k = 0; k < kin; ++k) {
      float xv = X[rc * (long)kin + k];
      v0 += xv * W1[k * 64 + c];
      v1 += xv * W1[k * 64 + c + 1];
    }
    v0 = fmaxf(v0, 0.f);
    v1 = fmaxf(v1, 0.f);
    int cl = c & 31;
    int ln = (r & 15) + (((cl & 15) >= 8) ? 16 : 0);
    int ii = ((cl >= 16) ? 8 : 0) + (cl & 7);
    int slot = (c >> 5) * 16 + ii;       // even
    bf16x2 pr; pr.x = (bf16)v0; pr.y = (bf16)v1;
    *(bf16x2*)&sA[r >> 4][ln][slot] = pr;
  }
  __syncthreads();
  f32x8 acc = {0.f,0.f,0.f,0.f,0.f,0.f,0.f,0.f};
#pragma unroll
  for (int s = 0; s < 2; ++s) {
    bf16x16 fa = *(const bf16x16*)&sA[rowsub][lane][s * 16];
    bf16x16 fb = ((const bf16x16*)W2f)[(s * 4 + colsub) * 32 + lane];
    acc = wmma_bf16f32(fa, fb, acc);
  }
  int ncol = colsub * 16 + (lane & 15);
  int mb = rowsub * 16 + ((lane < 16) ? 0 : 8);
  float bias = B2[ncol < nout ? ncol : 0];
#pragma unroll
  for (int r = 0; r < 8; ++r) {
    long row = rowbase + mb + r;
    if (row < M && ncol < nout)
      Y[row * (long)ldy + coff + ncol] = actf(acc[r] + bias, act);
  }
}

// ------------------------------ batch norm ------------------------------
__global__ void k_bn_stats(const float* __restrict__ X, long M, int ncols,
                           float* __restrict__ stats)
{
  __shared__ float ssum[64], ssq[64];
  int t = threadIdx.x;
  if (t < 64) { ssum[t] = 0.f; ssq[t] = 0.f; }
  __syncthreads();
  int c = t & 63;
  long r = (long)blockIdx.x * 4 + (t >> 6);
  long stride = (long)gridDim.x * 4;
  float s = 0.f, q = 0.f;
  for (; r < M; r += stride) {
    float v = X[r * 64 + c];
    s += v; q += v * v;
  }
  atomicAdd(&ssum[c], s);
  atomicAdd(&ssq[c], q);
  __syncthreads();
  if (t < 64 && t < ncols) {
    atomicAdd(&stats[t], ssum[t]);
    atomicAdd(&stats[64 + t], ssq[t]);
  }
}

__global__ void k_bn_apply(const float* __restrict__ Xs, long M, int ncols,
                           const float* __restrict__ st, const float* __restrict__ g,
                           const float* __restrict__ be, float invM, int dorelu,
                           float* __restrict__ Y, int ldy, int coff)
{
  long n = M * 64;
  long i = (long)blockIdx.x * blockDim.x + threadIdx.x;
  long stp = (long)gridDim.x * blockDim.x;
  for (; i < n; i += stp) {
    int c = (int)(i & 63);
    long r = i >> 6;
    if (c < ncols) {
      float m = st[c] * invM;
      float var = fmaxf(st[64 + c] * invM - m * m, 0.f);
      float o = (Xs[i] - m) * rsqrtf(var + 1e-5f) * g[c] + be[c];
      if (dorelu) o = fmaxf(o, 0.f);
      Y[r * (long)ldy + coff + c] = o;
    }
  }
}

// ------------------------------ graph ops ------------------------------
// out is [M,96], zero-padded once at launch; writes cols 0..64
__global__ void k_concat(const float* __restrict__ Xm, int ldx,
                         const float* __restrict__ s, float* __restrict__ out, long M)
{
  long n = M * 65;
  long i = (long)blockIdx.x * blockDim.x + threadIdx.x;
  long stp = (long)gridDim.x * blockDim.x;
  for (; i < n; i += stp) {
    long r = i / 65; int c = (int)(i - r * 65);
    out[r * 96 + c] = (c < 64) ? Xm[r * (long)ldx + c] : s[r];
  }
}

__global__ void k_setcol(float* __restrict__ m, int ld, int col,
                         const float* __restrict__ v, long M)
{
  long i = (long)blockIdx.x * blockDim.x + threadIdx.x;
  long stp = (long)gridDim.x * blockDim.x;
  for (; i < M; i += stp) m[i * (long)ld + col] = v[i];
}

// agg[dst[e]] += relu(node[src[e]] + ee[e]); 32 threads/edge, float2 each
__global__ void k_scatter(const float* __restrict__ node, const float* __restrict__ ee,
                          const int* __restrict__ src, const int* __restrict__ dst,
                          float* __restrict__ agg, long E)
{
  long n = E * 32;
  long i = (long)blockIdx.x * blockDim.x + threadIdx.x;
  long stp = (long)gridDim.x * blockDim.x;
  for (; i < n; i += stp) {
    long e = i >> 5; int c = (int)(i & 31) * 2;
    long s = src[e], d = dst[e];
    float2 nv = *(const float2*)&node[s * 64 + c];
    float2 ev = *(const float2*)&ee[e * 64 + c];
    float v0 = fmaxf(nv.x + ev.x, 0.f);
    float v1 = fmaxf(nv.y + ev.y, 0.f);
    atomicAdd(&agg[d * 64 + c], v0);
    atomicAdd(&agg[d * 64 + c + 1], v1);
  }
}

__global__ void k_err_init(const float* __restrict__ rhs, float* __restrict__ err, long M) {
  long i = (long)blockIdx.x * blockDim.x + threadIdx.x;
  long stp = (long)gridDim.x * blockDim.x;
  for (; i < M; i += stp) err[i] = -rhs[i];
}

__global__ void k_err_scatter(const float* __restrict__ a, const float* __restrict__ ef,
                              const int* __restrict__ src, const int* __restrict__ dst,
                              float* __restrict__ err, long E)
{
  long i = (long)blockIdx.x * blockDim.x + threadIdx.x;
  long stp = (long)gridDim.x * blockDim.x;
  for (; i < E; i += stp) atomicAdd(&err[dst[i]], a[src[i]] * ef[i]);
}

// agg = (1+eps)*state + agg   (state row stride lds_)
__global__ void k_residual(const float* __restrict__ state, int lds_,
                           const float* __restrict__ epsp, float* __restrict__ agg, long M)
{
  float s = 1.f + epsp[0];
  long n = M * 64;
  long i = (long)blockIdx.x * blockDim.x + threadIdx.x;
  long stp = (long)gridDim.x * blockDim.x;
  for (; i < n; i += stp) {
    long r = i >> 6; int c = (int)(i & 63);
    agg[i] = s * state[r * (long)lds_ + c] + agg[i];
  }
}

__global__ void k_logsm2(float* __restrict__ y, long M) {
  long i = (long)blockIdx.x * blockDim.x + threadIdx.x;
  long stp = (long)gridDim.x * blockDim.x;
  for (; i < M; i += stp) {
    float a = y[i * 2], b = y[i * 2 + 1];
    float m = fmaxf(a, b);
    float l = m + logf(expf(a - m) + expf(b - m));
    y[i * 2] = a - l; y[i * 2 + 1] = b - l;
  }
}

// ------------------------------ host side ------------------------------
namespace {
struct Cur {
  void* const* din; int idx; const float* blob; size_t off; bool leaf;
  const float* take(size_t n) {
    if (leaf) return (const float*)din[idx++];
    const float* p = blob + off; off += n; return p;
  }
};
struct MLPN { const float *b1, *b2, *w1, *w2; };
struct MLPB { const float *b1, *b2, *be, *g, *w1, *w2; };
struct BIPP { MLPB edge; const float* eps; MLPB joint; MLPB mlp; };
struct LINP { const float *b, *w; };
// jax tree_flatten order: dict keys sorted alphabetically, recursively.
MLPN takeN(Cur& c, int din_, int dmid, int dout) {
  MLPN m; m.b1 = c.take(dmid); m.b2 = c.take(dout);
  m.w1 = c.take((size_t)din_ * dmid); m.w2 = c.take((size_t)dmid * dout); return m;
}
MLPB takeB(Cur& c, int din_, int dmid, int dout) {
  MLPB m; m.b1 = c.take(dmid); m.b2 = c.take(dout); m.be = c.take(dout); m.g = c.take(dout);
  m.w1 = c.take((size_t)din_ * dmid); m.w2 = c.take((size_t)dmid * dout); return m;
}
BIPP takeBip(Cur& c, int jo) {  // keys: edge < eps < joint < mlp
  BIPP b; b.edge = takeB(c, 1, 64, 64); b.eps = c.take(1);
  b.joint = takeB(c, 65, 64, jo); b.mlp = takeB(c, 64, 64, 64); return b;
}
LINP takeL(Cur& c, int din_, int dout) { LINP l; l.b = c.take(dout); l.w = c.take((size_t)din_ * dout); return l; }
}

extern "C" void kernel_launch(void* const* d_in, const int* in_sizes, int n_in,
                              void* d_out, int out_size, void* d_ws, size_t ws_size,
                              hipStream_t stream)
{
  (void)in_sizes; (void)out_size; (void)ws_size;
  const float* vnf = (const float*)d_in[0];
  const float* cnf = (const float*)d_in[1];
  const float* efv = (const float*)d_in[2];
  const float* efc = (const float*)d_in[3];
  const float* rhs = (const float*)d_in[4];

  Cur cur; cur.din = d_in; cur.idx = 5; cur.off = 0;
  cur.leaf = (n_in > 16);
  cur.blob = cur.leaf ? nullptr : (const float*)d_in[5];

  // top-level sorted: assign_1..4, con_enc, cv1..4, lin1..4, var_enc, vc1..4
  MLPN assign[4]; for (int i = 0; i < 4; ++i) assign[i] = takeN(cur, 64, 64, 1);
  MLPN con_enc = takeN(cur, 2, 64, 64);
  BIPP cv[4]; for (int i = 0; i < 4; ++i) cv[i] = takeBip(cur, 63);
  LINP lin1 = takeL(cur, 320, 64), lin2 = takeL(cur, 64, 64);
  LINP lin3 = takeL(cur, 64, 64),  lin4 = takeL(cur, 64, 2);
  MLPN var_enc = takeN(cur, 2, 64, 64);
  BIPP vc[4]; for (int i = 0; i < 4; ++i) vc[i] = takeBip(cur, 64);

  const int *eiv, *eic;
  if (cur.leaf) { eiv = (const int*)d_in[cur.idx]; eic = (const int*)d_in[cur.idx + 1]; }
  else          { eiv = (const int*)d_in[6];       eic = (const int*)d_in[7]; }
  const int* sv = eiv;        const int* tv = eiv + Ee;
  const int* sc = eic;        const int* tc = eic + Ee;

  // workspace layout (floats)
  float* p = (float*)d_ws;
  auto alloc = [&](long n) { float* q = p; p += n; return q; };
  float* EE   = alloc(Ee * 64);       // edge activations [E,64]
  float* NODE = alloc(NVv * 64);      // ns / jc / mlp out
  float* AGG  = alloc(NVv * 64);      // segment sums / residual input
  float* HT   = alloc(NVv * 64);      // MLP mid layer
  float* C    = alloc(NCc * 64);      // constraint state
  float* Xb   = alloc(NVv * 320);     // vs concat [NV,320]
  float* CAT  = alloc(NVv * 96);      // joint input, zero-padded K=96
  float* A    = alloc(NVv);
  float* ERR  = alloc(NCc);
  float* ST   = alloc(128);
  bf16* frag  = (bf16*)alloc(300000); // bf16 weight-fragment arena (1.2MB)
  bf16* fp_   = frag;

  auto blks = [](long n) { long b = (n + 255) / 256; if (b > 1048576) b = 1048576; return (unsigned)b; };
  // pre-swizzle a weight matrix into padded B-fragment layout
  auto prep = [&](const float* W, int K, int nout) -> const bf16* {
    int Kpad = (K + 31) & ~31;
    bf16* F = fp_; fp_ += (long)Kpad * 64;
    k_prep_w<<<blks((long)Kpad * 64), 256, 0, stream>>>(W, K, nout, F, Kpad);
    return F;
  };
  auto gemm = [&](const float* X, int ldx, long M, int K, const bf16* Wf, const float* B,
                  int nout, float* Y, int ldy, int coff, int act) {
    unsigned g = (unsigned)((M + 31) / 32);
    if (K == 64)       k_gemm_t<64><<<g, 256, 0, stream>>>(X, ldx, M, Wf, B, nout, Y, ldy, coff, act);
    else if (K == 96)  k_gemm_t<96><<<g, 256, 0, stream>>>(X, ldx, M, Wf, B, nout, Y, ldy, coff, act);
    else               k_gemm_t<320><<<g, 256, 0, stream>>>(X, ldx, M, Wf, B, nout, Y, ldy, coff, act);
  };
  auto bn = [&](float* src, long M, int ncols, const float* g, const float* be,
                int dorelu, float* dst, int ldy, int coff) {
    k_zero<<<1, 128, 0, stream>>>(ST, 128);
    k_bn_stats<<<512, 256, 0, stream>>>(src, M, ncols, ST);
    k_bn_apply<<<blks(M * 64), 256, 0, stream>>>(src, M, ncols, ST, g, be, 1.f / (float)M, dorelu, dst, ldy, coff);
  };

  // ---- prep all GEMM weights once (tiny; stays hot in L2) ----
  const bf16 *f_venc = prep(var_enc.w2, 64, 64), *f_cenc = prep(con_enc.w2, 64, 64);
  const bf16 *f_as1[4], *f_as2[4];
  const bf16 *f_vc_e[4], *f_vc_j1[4], *f_vc_j2[4], *f_vc_m1[4], *f_vc_m2[4];
  const bf16 *f_cv_e[4], *f_cv_j1[4], *f_cv_j2[4], *f_cv_m1[4], *f_cv_m2[4];
  for (int i = 0; i < 4; ++i) {
    f_as1[i] = prep(assign[i].w1, 64, 64);
    f_as2[i] = prep(assign[i].w2, 64, 1);
    f_vc_e[i]  = prep(vc[i].edge.w2, 64, 64);
    f_vc_j1[i] = prep(vc[i].joint.w1, 65, 64);   // padded to K=96
    f_vc_j2[i] = prep(vc[i].joint.w2, 64, 64);
    f_vc_m1[i] = prep(vc[i].mlp.w1, 64, 64);
    f_vc_m2[i] = prep(vc[i].mlp.w2, 64, 64);
    f_cv_e[i]  = prep(cv[i].edge.w2, 64, 64);
    f_cv_j1[i] = prep(cv[i].joint.w1, 65, 64);
    f_cv_j2[i] = prep(cv[i].joint.w2, 64, 63);
    f_cv_m1[i] = prep(cv[i].mlp.w1, 64, 64);
    f_cv_m2[i] = prep(cv[i].mlp.w2, 64, 64);
  }
  const bf16 *f_l1 = prep(lin1.w, 320, 64), *f_l2 = prep(lin2.w, 64, 64);
  const bf16 *f_l3 = prep(lin3.w, 64, 64),  *f_l4 = prep(lin4.w, 64, 2);

  k_zero<<<blks(NVv * 96), 256, 0, stream>>>(CAT, NVv * 96);  // pad cols stay 0

  // encoders: v -> Xb slice 0 (ld 320), c -> C
  k_fused_mlp2<<<(unsigned)((NVv + 31) / 32), 256, 0, stream>>>(
      vnf, 2, NVv, var_enc.w1, var_enc.b1, f_venc, var_enc.b2, 64, Xb, 320, 0, 0);
  k_fused_mlp2<<<(unsigned)((NCc + 31) / 32), 256, 0, stream>>>(
      cnf, 2, NCc, con_enc.w1, con_enc.b1, f_cenc, con_enc.b2, 64, C, 64, 0, 0);

  for (int i = 0; i < 4; ++i) {
    const float* vprev = Xb + 64 * i;  // row stride 320
    BIPP& V = vc[i]; BIPP& Q = cv[i]; MLPN& As = assign[i];

    // ---------- var_con ----------
    k_fused_mlp2<<<(unsigned)((Ee + 31) / 32), 256, 0, stream>>>(
        efv, 1, Ee, V.edge.w1, V.edge.b1, f_vc_e[i], V.edge.b2, 64, EE, 64, 0, 1);
    bn(EE, Ee, 64, V.edge.g, V.edge.be, 0, EE, 64, 0);
    gemm(vprev, 320, NVv, 64, f_as1[i], As.b1, 64, HT, 64, 0, 1);
    gemm(HT, 64, NVv, 64, f_as2[i], As.b2, 1, A, 1, 0, 2);
    k_concat<<<blks(NVv * 65), 256, 0, stream>>>(vprev, 320, A, CAT, NVv);
    gemm(CAT, 96, NVv, 96, f_vc_j1[i], V.joint.b1, 64, HT, 64, 0, 1);
    gemm(HT, 64, NVv, 64, f_vc_j2[i], V.joint.b2, 64, NODE, 64, 0, 1);
    bn(NODE, NVv, 64, V.joint.g, V.joint.be, 0, NODE, 64, 0);
    k_zero<<<blks(NCc * 64), 256, 0, stream>>>(AGG, NCc * 64);
    k_scatter<<<blks(Ee * 32), 256, 0, stream>>>(NODE, EE, sv, tv, AGG, Ee);
    k_residual<<<blks(NCc * 64), 256, 0, stream>>>(C, 64, V.eps, AGG, NCc);
    gemm(AGG, 64, NCc, 64, f_vc_m1[i], V.mlp.b1, 64, HT, 64, 0, 1);
    gemm(HT, 64, NCc, 64, f_vc_m2[i], V.mlp.b2, 64, NODE, 64, 0, 1);
    bn(NODE, NCc, 64, V.mlp.g, V.mlp.be, 1, C, 64, 0);   // round-level relu

    // ---------- error (always assign_1 params, current v) ----------
    gemm(vprev, 320, NVv, 64, f_as1[0], assign[0].b1, 64, HT, 64, 0, 1);
    gemm(HT, 64, NVv, 64, f_as2[0], assign[0].b2, 1, A, 1, 0, 2);
    k_err_init<<<blks(NCc), 256, 0, stream>>>(rhs, ERR, NCc);
    k_err_scatter<<<blks(Ee), 256, 0, stream>>>(A, efv, sv, tv, ERR, Ee);

    // ---------- con_var ----------
    k_fused_mlp2<<<(unsigned)((Ee + 31) / 32), 256, 0, stream>>>(
        efc, 1, Ee, Q.edge.w1, Q.edge.b1, f_cv_e[i], Q.edge.b2, 64, EE, 64, 0, 1);
    bn(EE, Ee, 64, Q.edge.g, Q.edge.be, 0, EE, 64, 0);
    k_concat<<<blks(NCc * 65), 256, 0, stream>>>(C, 64, ERR, CAT, NCc);
    gemm(CAT, 96, NCc, 96, f_cv_j1[i], Q.joint.b1, 64, HT, 64, 0, 1);
    gemm(HT, 64, NCc, 64, f_cv_j2[i], Q.joint.b2, 63, NODE, 64, 0, 1);
    bn(NODE, NCc, 63, Q.joint.g, Q.joint.be, 0, NODE, 64, 0);
    k_setcol<<<blks(NCc), 256, 0, stream>>>(NODE, 64, 63, ERR, NCc);  // jc = [joint, err]
    k_zero<<<blks(NVv * 64), 256, 0, stream>>>(AGG, NVv * 64);
    k_scatter<<<blks(Ee * 32), 256, 0, stream>>>(NODE, EE, sc, tc, AGG, Ee);
    k_residual<<<blks(NVv * 64), 256, 0, stream>>>(vprev, 320, Q.eps, AGG, NVv);
    gemm(AGG, 64, NVv, 64, f_cv_m1[i], Q.mlp.b1, 64, HT, 64, 0, 1);
    gemm(HT, 64, NVv, 64, f_cv_m2[i], Q.mlp.b2, 64, NODE, 64, 0, 1);
    bn(NODE, NVv, 64, Q.mlp.g, Q.mlp.be, 1, Xb, 320, 64 * (i + 1));  // v_{i+1}
  }

  // ---------- head ----------
  gemm(Xb, 320, NVv, 320, f_l1, lin1.b, 64, HT, 64, 0, 1);
  gemm(HT, 64, NVv, 64, f_l2, lin2.b, 64, AGG, 64, 0, 1);
  gemm(AGG, 64, NVv, 64, f_l3, lin3.b, 64, NODE, 64, 0, 1);
  gemm(NODE, 64, NVv, 64, f_l4, lin4.b, 2, (float*)d_out, 2, 0, 0);
  k_logsm2<<<blks(NVv), 256, 0, stream>>>((float*)d_out, NVv);
}
